// YiJingAttention_75754633167493
// MI455X (gfx1250) — matrix operand order repaired and checked by reading, not compiled, not run on hardware
//
#include <hip/hip_runtime.h>
#include <hip/hip_bf16.h>

typedef __attribute__((ext_vector_type(16))) _Float16 v16h;
typedef __attribute__((ext_vector_type(8)))  float    v8f;
typedef __attribute__((ext_vector_type(4)))  unsigned int v4u;

union FragU { v16h v; v4u u[2]; _Float16 e[16]; };

// A-fragment: 16x32 f16 tile, row-major global source; lane holds row (lane&15).
// lanes 0-15 -> K = 0..15 contiguous; lanes 16-31 -> K = 16..31.
__device__ __forceinline__ v16h ld_frag_a(const _Float16* src, int ld, int rowbase,
                                          int kbase, int lane) {
  int r = lane & 15, half = lane >> 4;
  const _Float16* p = src + (size_t)(rowbase + r) * ld + kbase + half * 16;
  FragU f;
  f.u[0] = *reinterpret_cast<const v4u*>(p);
  f.u[1] = *reinterpret_cast<const v4u*>(p + 8);
  return f.v;
}

// B-fragment: 32x16 f16 tile; srcT is B transposed (row n = column n of B).
// lanes 0-15 -> K=0..15 contiguous; lanes 16-31 -> K=16..31.
__device__ __forceinline__ v16h ld_frag_b(const _Float16* srcT, int ld, int colbase,
                                          int kbase, int lane) {
  int n = lane & 15, half = lane >> 4;
  const _Float16* p = srcT + (size_t)(colbase + n) * ld + kbase + half * 16;
  FragU f;
  f.u[0] = *reinterpret_cast<const v4u*>(p);
  f.u[1] = *reinterpret_cast<const v4u*>(p + 8);
  return f.v;
}

// A-fragment from LDS-staged 64x32 f16 macro-tile (row pitch 80 B, 16B-aligned,
// bank-conflict free: (20*r) mod 64 covers 16 distinct banks).
__device__ __forceinline__ v16h ld_frag_a_lds(const char* lds, int mt, int lane) {
  int r = lane & 15, half = lane >> 4;
  const char* p = lds + (mt * 16 + r) * 80 + half * 32;
  FragU f;
  f.u[0] = *reinterpret_cast<const v4u*>(p);       // cols (elem) half*16 + 0..7
  f.u[1] = *reinterpret_cast<const v4u*>(p + 16);  // cols half*16 + 8..15
  return f.v;
}

__device__ __forceinline__ v8f wmma16(v16h a, v16h b, v8f c) {
  return __builtin_amdgcn_wmma_f32_16x16x32_f16(false, a, false, b, (short)0, c,
                                                false, false);
}

// Cooperative async copy: 64 rows x 64 B of X into LDS (pitch 80B), 128 threads,
// 2 x b128 per thread, tracked by ASYNCcnt (gfx1250 async global->LDS engine).
__device__ __forceinline__ void async_copy_a(const _Float16* X, char* lds,
                                             int m0, int k, int tid) {
#pragma unroll
  for (int i = 0; i < 2; ++i) {
    int c = tid + 128 * i;            // chunk id 0..255
    int row = c >> 2;
    int cb = (c & 3) * 16;            // byte offset within 64B row
    unsigned laddr = (unsigned)(uintptr_t)(lds + row * 80 + cb);
    const char* g = (const char*)X + ((size_t)(m0 + row) * 1024 + k) * 2 + cb;
    asm volatile("global_load_async_to_lds_b128 %0, %1, off"
                 :: "v"(laddr), "v"(g) : "memory");
  }
}

__device__ __forceinline__ void wait_async0() {
  asm volatile("s_wait_asynccnt 0x0" ::: "memory");
}

// ---------------- precision conversion / transpose ----------------

__global__ void k_cvt_f16(const float* __restrict__ in, _Float16* __restrict__ out, int n) {
  int i = blockIdx.x * blockDim.x + threadIdx.x;
  if (i < n) out[i] = (_Float16)in[i];
}

// out[cols_in, rows_in] = in^T (coalesced writes)
__global__ void k_transpose_f16(const float* __restrict__ in, _Float16* __restrict__ out,
                                int rows_in, int cols_in) {
  int i = blockIdx.x * blockDim.x + threadIdx.x;
  if (i >= rows_in * cols_in) return;
  int j = i / rows_in;
  int r = i - j * rows_in;
  out[i] = (_Float16)in[(size_t)r * cols_in + j];
}

// ---------------- QKV projection GEMM (64x32 per wave, LDS-staged A) ----------------
__global__ void __launch_bounds__(128, 1)
k_qkv_gemm(const _Float16* __restrict__ X, const _Float16* __restrict__ WT,
           _Float16* __restrict__ qb, _Float16* __restrict__ kb,
           _Float16* __restrict__ vt) {
  __shared__ __align__(16) char As[2][64 * 80];
  int tid = threadIdx.x;
  int lane = tid & 31;
  int wave = tid >> 5;
  int mb = blockIdx.x / 24;               // 64 M-blocks of 64 rows
  int nb = blockIdx.x % 24;               // 24 N-macros of 128 cols
  int m0 = mb * 64;
  int n0 = nb * 128 + wave * 32;          // this wave: 64x32 tile

  v8f acc[4][2];
  v8f zero = {0.f, 0.f, 0.f, 0.f, 0.f, 0.f, 0.f, 0.f};
#pragma unroll
  for (int i = 0; i < 4; ++i) { acc[i][0] = zero; acc[i][1] = zero; }

  async_copy_a(X, As[0], m0, 0, tid);

  for (int ks = 0; ks < 32; ++ks) {
    int k = ks * 32;
    wait_async0();
    __syncthreads();
    const char* ab = As[ks & 1];
    if (ks + 1 < 32) async_copy_a(X, As[(ks + 1) & 1], m0, k + 32, tid);
    if (k + 64 < 1024)
      __builtin_prefetch(WT + (size_t)(n0 + (lane & 15)) * 1024 + k + 64, 0, 0);

    v16h b0 = ld_frag_b(WT, 1024, n0 + 0,  k, lane);
    v16h b1 = ld_frag_b(WT, 1024, n0 + 16, k, lane);
#pragma unroll
    for (int mt = 0; mt < 4; ++mt) {
      v16h a = ld_frag_a_lds(ab, mt, lane);
      acc[mt][0] = wmma16(a, b0, acc[mt][0]);
      acc[mt][1] = wmma16(a, b1, acc[mt][1]);
    }
  }

  int col = lane & 15, half = lane >> 4;
#pragma unroll
  for (int nt = 0; nt < 2; ++nt) {
    int n0s = n0 + nt * 16;
    int s = n0s >> 10;                    // 0=q 1=k 2=v
    int h = (n0s >> 7) & 7;
    int d = (n0s & 127) + col;
#pragma unroll
    for (int mt = 0; mt < 4; ++mt) {
#pragma unroll
      for (int r = 0; r < 8; ++r) {
        int m = m0 + mt * 16 + r + 8 * half;
        int b_ = m >> 11, t = m & 2047;
        _Float16 val = (_Float16)acc[mt][nt][r];
        size_t bh = (size_t)((b_ << 3) + h);
        if (s == 0)      qb[(bh * 2048 + t) * 128 + d] = val;
        else if (s == 1) kb[(bh * 2048 + t) * 128 + d] = val;
        else             vt[(bh * 128 + d) * 2048 + t] = val;
      }
    }
  }
}

// ---------------- fused flash attention with trigram geo bias ----------------
__global__ void __launch_bounds__(128, 1)
k_attn(const _Float16* __restrict__ qbuf, const _Float16* __restrict__ kbuf,
       const _Float16* __restrict__ vtb, const float* __restrict__ hscales,
       _Float16* __restrict__ ob) {
  int lane = threadIdx.x & 31;
  int idx = blockIdx.x * 4 + (threadIdx.x >> 5);
  int qt = idx & 127;
  int bh = idx >> 7;
  int h = bh & 7;
  int b = bh >> 3;
  const _Float16* qh = qbuf + (size_t)bh * 2048 * 128;
  const _Float16* kh = kbuf + (size_t)bh * 2048 * 128;
  const _Float16* vh = vtb + (size_t)bh * 128 * 2048;
  int qbase = qt * 16;
  int qcol = lane & 15, half = lane >> 4;
  int qglob = qbase + qcol;
  const float rs3 = 0.57735026919f;
  float dx = (float)(((h >> 2) & 1) * 2 - 1) * rs3;
  float dy = (float)(((h >> 1) & 1) * 2 - 1) * rs3;
  float dz = (float)((h & 1) * 2 - 1) * rs3;
  float hs = hscales[h];
  const _Float16* qr = qh + (size_t)qglob * 128;
  float qp = (float)qr[0] * dx + (float)qr[1] * dy + (float)qr[2] * dz;
  const float scale = 0.088388347648318447f;   // 1/sqrt(128)

  v8f o[8];
  v8f zero = {0.f, 0.f, 0.f, 0.f, 0.f, 0.f, 0.f, 0.f};
#pragma unroll
  for (int i = 0; i < 8; ++i) o[i] = zero;
  float mrun = -1e30f, lrun = 0.0f;

  for (int kc = 0; kc < qbase + 16; kc += 32) {
    v8f s0 = zero, s1 = zero;
#pragma unroll
    for (int cb = 0; cb < 128; cb += 32) {
      v16h bq = ld_frag_b(qh, 128, qbase, cb, lane);
      v16h a0 = ld_frag_a(kh, 128, kc, cb, lane);
      v16h a1 = ld_frag_a(kh, 128, kc + 16, cb, lane);
      s0 = wmma16(a0, bq, s0);
      s1 = wmma16(a1, bq, s1);
    }
    const _Float16* krw = kh + (size_t)(kc + lane) * 128;
    float kpl = (float)krw[0] * dx + (float)krw[1] * dy + (float)krw[2] * dz;

    float e0[8], e1[8];
    float cmax = -1e30f;
#pragma unroll
    for (int r = 0; r < 8; ++r) {
      int kg = kc + r + 8 * half;
      float kp0 = __shfl(kpl, r + 8 * half, 32);
      float kp1 = __shfl(kpl, 16 + r + 8 * half, 32);
      float v0 = s0[r] * scale + hs * qp * kp0;
      float v1 = s1[r] * scale + hs * qp * kp1;
      v0 = (kg > qglob) ? -1e30f : v0;
      v1 = (kg + 16 > qglob) ? -1e30f : v1;
      e0[r] = v0; e1[r] = v1;
      cmax = fmaxf(cmax, fmaxf(v0, v1));
    }
    cmax = fmaxf(cmax, __shfl_xor(cmax, 16, 32));
    float mnew = fmaxf(mrun, cmax);
    float resc = __expf(mrun - mnew);
    float p0[8], p1[8], psum = 0.f;
#pragma unroll
    for (int r = 0; r < 8; ++r) {
      p0[r] = __expf(e0[r] - mnew);
      p1[r] = __expf(e1[r] - mnew);
      psum += p0[r] + p1[r];
    }
    psum += __shfl_xor(psum, 16, 32);
    lrun = lrun * resc + psum;
    mrun = mnew;

    FragU pb;
#pragma unroll
    for (int r = 0; r < 8; ++r) {
      float q0 = __shfl_xor(p0[r], 16, 32);
      float q1 = __shfl_xor(p1[r], 16, 32);
      float lo = half ? q1 : p0[r];
      float hi = half ? p1[r] : q0;
      pb.e[r] = (_Float16)lo;
      pb.e[8 + r] = (_Float16)hi;
    }
#pragma unroll
    for (int dt = 0; dt < 8; ++dt) {
#pragma unroll
      for (int r = 0; r < 8; ++r) o[dt][r] *= resc;
      v16h av = ld_frag_a(vh, 2048, dt * 16, kc, lane);
      o[dt] = wmma16(av, pb.v, o[dt]);
    }
  }

  float linv = 1.0f / lrun;
  size_t rowbase = ((size_t)b * 2048 + qglob) * 1024 + h * 128;
#pragma unroll
  for (int dt = 0; dt < 8; ++dt)
#pragma unroll
    for (int r = 0; r < 8; ++r)
      ob[rowbase + dt * 16 + r + 8 * half] = (_Float16)(o[dt][r] * linv);
}

// ---------------- output projection GEMM (64x32 per wave, LDS-staged A) ----------------
__global__ void __launch_bounds__(128, 1)
k_out_gemm(const _Float16* __restrict__ O, const _Float16* __restrict__ WT,
           float* __restrict__ Y) {
  __shared__ __align__(16) char As[2][64 * 80];
  int tid = threadIdx.x;
  int lane = tid & 31;
  int wave = tid >> 5;
  int mb = blockIdx.x >> 3;               // 64 M-blocks
  int nb = blockIdx.x & 7;                // 8 N-macros of 128
  int m0 = mb * 64;
  int n0 = nb * 128 + wave * 32;

  v8f acc[4][2];
  v8f zero = {0.f, 0.f, 0.f, 0.f, 0.f, 0.f, 0.f, 0.f};
#pragma unroll
  for (int i = 0; i < 4; ++i) { acc[i][0] = zero; acc[i][1] = zero; }

  async_copy_a(O, As[0], m0, 0, tid);

  for (int ks = 0; ks < 32; ++ks) {
    int k = ks * 32;
    wait_async0();
    __syncthreads();
    const char* ab = As[ks & 1];
    if (ks + 1 < 32) async_copy_a(O, As[(ks + 1) & 1], m0, k + 32, tid);
    if (k + 64 < 1024)
      __builtin_prefetch(WT + (size_t)(n0 + (lane & 15)) * 1024 + k + 64, 0, 0);

    v16h b0 = ld_frag_b(WT, 1024, n0 + 0,  k, lane);
    v16h b1 = ld_frag_b(WT, 1024, n0 + 16, k, lane);
#pragma unroll
    for (int mt = 0; mt < 4; ++mt) {
      v16h a = ld_frag_a_lds(ab, mt, lane);
      acc[mt][0] = wmma16(a, b0, acc[mt][0]);
      acc[mt][1] = wmma16(a, b1, acc[mt][1]);
    }
  }

  int col = lane & 15, half = lane >> 4;
#pragma unroll
  for (int nt = 0; nt < 2; ++nt) {
    int n = n0 + nt * 16 + col;
#pragma unroll
    for (int mt = 0; mt < 4; ++mt) {
#pragma unroll
      for (int r = 0; r < 8; ++r) {
        int m = m0 + mt * 16 + r + 8 * half;
        Y[(size_t)m * 1024 + n] = acc[mt][nt][r];
      }
    }
  }
}

extern "C" void kernel_launch(void* const* d_in, const int* in_sizes, int n_in,
                              void* d_out, int out_size, void* d_ws, size_t ws_size,
                              hipStream_t stream) {
  const float* x    = (const float*)d_in[0];   // [2,2048,1024]
  const float* wqkv = (const float*)d_in[1];   // [1024,3072]
  const float* wout = (const float*)d_in[2];   // [1024,1024]
  const float* hsc  = (const float*)d_in[3];   // [8]
  float* y = (float*)d_out;

  char* w = (char*)d_ws;
  const size_t MB = 1024 * 1024;
  _Float16* xh  = (_Float16*)(w + 0 * MB);   // 8 MiB  x f16 [4096,1024]
  _Float16* wqT = (_Float16*)(w + 8 * MB);   // 6 MiB  wqkv^T [3072,1024]
  _Float16* woT = (_Float16*)(w + 14 * MB);  // 2 MiB  wout^T [1024,1024]
  _Float16* qb  = (_Float16*)(w + 16 * MB);  // 8 MiB  q [B,H,T,128]
  _Float16* kb  = (_Float16*)(w + 24 * MB);  // 8 MiB  k [B,H,T,128]
  _Float16* vt  = (_Float16*)(w + 32 * MB);  // 8 MiB  v^T [B,H,128,T]
  _Float16* ob  = (_Float16*)(w + 40 * MB);  // 8 MiB  attn out [4096,1024]

  k_cvt_f16<<<4194304 / 256, 256, 0, stream>>>(x, xh, 4194304);
  k_transpose_f16<<<3145728 / 256, 256, 0, stream>>>(wqkv, wqT, 1024, 3072);
  k_transpose_f16<<<1048576 / 256, 256, 0, stream>>>(wout, woT, 1024, 1024);

  // 64 M-blocks x 24 N-macros; 4 waves/block, each wave 64x32 of output
  k_qkv_gemm<<<64 * 24, 128, 0, stream>>>(xh, wqT, qb, kb, vt);

  // 2*8*128 q-tiles, 4 waves per block
  k_attn<<<2048 / 4, 128, 0, stream>>>(qb, kb, vt, hsc, ob);

  // 64 M-blocks x 8 N-macros
  k_out_gemm<<<64 * 8, 128, 0, stream>>>(ob, woT, y);
}
